// modelLSTMElec_64510408786512
// MI455X (gfx1250) — matrix operand order repaired and checked by reading, not compiled, not run on hardware
//
#include <hip/hip_runtime.h>

#define T_LEN 16384
#define D_IN  14
#define H_DIM 32
#define M_N   4
#define G_DIM 128   // 4*H

typedef __attribute__((ext_vector_type(16))) _Float16 v16h;
typedef __attribute__((ext_vector_type(8)))  _Float16 v8h;
typedef __attribute__((ext_vector_type(8)))  float    v8f;

__device__ __forceinline__ float fast_sigmoid(float x) {
  // 1 / (1 + exp(-x)) via v_exp_f32 (base-2) + v_rcp_f32
  float e = __builtin_amdgcn_exp2f(-1.44269504088896340736f * x);
  return __builtin_amdgcn_rcpf(1.0f + e);
}
__device__ __forceinline__ float fast_tanh(float x) {
  x = fminf(fmaxf(x, -15.0f), 15.0f);              // keep exp2 finite
  float e = __builtin_amdgcn_exp2f(-2.88539008177792681472f * x); // exp(-2x)
  return (1.0f - e) * __builtin_amdgcn_rcpf(1.0f + e);
}

// ---------------------------------------------------------------------------
// Phase 1: P[m][t][g] = sum_d W_ih[m][g][d]*x[m][t][d] + b_ih[m][g] + b_hh[m][g]
// One wave per (m, 16-timestep block); 8 WMMAs per wave (gate tiles).
//   A: 16(t) x 32(d, zero-padded past 14), f16
//   B: 32(d) x 16(g), f16 = W_ih^T tile
//   D: 16(t) x 16(g), f32
// All padding done with branchless clamped loads + v_cndmask (no EXEC games).
// ---------------------------------------------------------------------------
__global__ __launch_bounds__(128) void lstm_xgate_precompute(
    const float* __restrict__ x, const float* __restrict__ W_ih,
    const float* __restrict__ b_ih, const float* __restrict__ b_hh,
    float* __restrict__ P)
{
  const int wave = (blockIdx.x * 128 + (int)threadIdx.x) >> 5;  // 0..4095
  const int lane = threadIdx.x & 31;
  const int m  = wave & 3;
  const int t0 = (wave >> 2) << 4;
  const bool lo = lane < 16;
  const int r  = lane & 15;          // A row (local timestep) / D column index
  const int kb = lo ? 0 : 8;         // K base for this lane group

  // A operand: lane holds row t0+r; element e -> K = kb + (e<8 ? e : e+8)
  const float* xrow = x + ((size_t)m * T_LEN + (size_t)(t0 + r)) * D_IN;
  v16h A;
#pragma unroll
  for (int e = 0; e < 16; ++e) {
    const int d  = kb + (e < 8 ? e : e + 8);
    const int dc = (d < D_IN) ? d : 0;            // clamped, always in-bounds
    const float v = xrow[dc];                     // unconditional load
    A[e] = (_Float16)((d < D_IN) ? v : 0.0f);     // v_cndmask, no branches
  }

  const bool lval = lane < D_IN;
  const int  lc   = lval ? lane : 0;              // clamped B row index
  const v8f zc = {0.f, 0.f, 0.f, 0.f, 0.f, 0.f, 0.f, 0.f};

#pragma unroll
  for (int tau = 0; tau < 8; ++tau) {
    const int g0 = tau << 4;
    // B operand: lane column = gate g0+r; element n -> K = kb + (n<8 ? n : n+8)
    v16h B;
#pragma unroll
    for (int n = 0; n < 16; ++n) {
      const int d = kb + (n < 8 ? n : n + 8);
      const float w = W_ih[((size_t)m * G_DIM + g0 + r) * D_IN + ((d < D_IN) ? d : 0)];
      B[n] = (_Float16)((d < D_IN) ? w : 0.0f);
    }

    v8f acc = __builtin_amdgcn_wmma_f32_16x16x32_f16(
        false, A, false, B, (short)0, zc, false, false);

    const float bs = b_ih[m * G_DIM + g0 + r] + b_hh[m * G_DIM + g0 + r];
    // D layout: VGPR p = rows p (lanes 0-15) and p+8 (lanes 16-31), N = r
#pragma unroll
    for (int p = 0; p < 8; ++p) {
      const int t = t0 + p + (lo ? 0 : 8);
      P[((size_t)m * T_LEN + t) * G_DIM + g0 + r] = acc[p] + bs;
    }
  }
}

// ---------------------------------------------------------------------------
// Phase 2: sequential recurrence. One workgroup, wave m owns LSTM m.
// Per step: gates = (rows-replicated h) x W_hh^T via 8 independent WMMAs.
// All A rows == h  =>  D VGPR0, lane L = gate[16*tau + L%16] in every lane.
// ---------------------------------------------------------------------------
__device__ __forceinline__ void lstm_step(const v16h (&Bt)[8],
                                          _Float16* __restrict__ hrow,
                                          const int kb, const bool lo,
                                          const int lane,
                                          const float (&pc)[8],
                                          float& h, float& c)
{
  // Build A: every row = h (f16). Lane needs h[kb..kb+7] and h[kb+16..kb+23]
  const v8h a0 = *(const v8h*)(hrow + kb);
  const v8h a1 = *(const v8h*)(hrow + kb + 16);
  const v16h A = __builtin_shufflevector(a0, a1, 0, 1, 2, 3, 4, 5, 6, 7, 8, 9,
                                         10, 11, 12, 13, 14, 15);
  const v8f zc = {0.f, 0.f, 0.f, 0.f, 0.f, 0.f, 0.f, 0.f};

  float g[8];
#pragma unroll
  for (int tau = 0; tau < 8; ++tau) {
    v8f acc = __builtin_amdgcn_wmma_f32_16x16x32_f16(
        false, A, false, Bt[tau], (short)0, zc, false, false);
    g[tau] = acc[0] + pc[tau];          // lane L holds gate 16*tau + L%16
  }

  // lane j owns hidden unit j; torch gate order i,f,g,o across tiles 0..7
  const float iv = lo ? g[0] : g[1];
  const float fv = lo ? g[2] : g[3];
  const float gv = lo ? g[4] : g[5];
  const float ov = lo ? g[6] : g[7];

  c = fast_sigmoid(fv) * c + fast_sigmoid(iv) * fast_tanh(gv);
  h = fast_sigmoid(ov) * fast_tanh(c);

  hrow[lane] = (_Float16)h;             // broadcast h for next step (LDS f16)
  __asm__ volatile("" ::: "memory");    // keep store before next step's loads
}

__global__ __launch_bounds__(128) void lstm_recurrence(
    const float* __restrict__ P, const float* __restrict__ W_hh,
    const float* __restrict__ Wt, const float* __restrict__ bt,
    const float* __restrict__ Wf1, const float* __restrict__ bf1,
    const float* __restrict__ Wf2, const float* __restrict__ bf2,
    float* __restrict__ out)
{
  __shared__ __align__(16) _Float16 hsh[M_N][H_DIM];
  __shared__ float hfin[M_N][H_DIM];

  const int m    = threadIdx.x >> 5;
  const int lane = threadIdx.x & 31;
  const bool lo  = lane < 16;
  const int r    = lane & 15;
  const int kb   = lo ? 0 : 8;

  // Preload B operands (register-resident): lane column = gate; element n -> k = kb + (n<8 ? n : n+8)
  v16h Bt[8];
#pragma unroll
  for (int tau = 0; tau < 8; ++tau)
#pragma unroll
    for (int n = 0; n < 16; ++n)
      Bt[tau][n] =
          (_Float16)W_hh[((size_t)m * G_DIM + tau * 16 + r) * H_DIM + kb + (n < 8 ? n : n + 8)];

  float h = 0.0f, c = 0.0f;
  _Float16* hrow = &hsh[m][0];
  hrow[lane] = (_Float16)0.0f;          // per-wave LDS row; in-order DS pipe
  __asm__ volatile("" ::: "memory");

  const float* Pm = P + (size_t)m * T_LEN * G_DIM;

  float pc[8], pn[8];
#pragma unroll
  for (int tau = 0; tau < 8; ++tau) pc[tau] = Pm[tau * 16 + r];

  // 2x-unrolled software pipeline: prefetch one step ahead, ping-pong pc/pn
  for (int t = 0; t < T_LEN; t += 2) {
#pragma unroll
    for (int tau = 0; tau < 8; ++tau)
      pn[tau] = Pm[(size_t)(t + 1) * G_DIM + tau * 16 + r];

    lstm_step(Bt, hrow, kb, lo, lane, pc, h, c);          // step t

    const int t2 = (t + 2 < T_LEN) ? t + 2 : T_LEN - 1;   // clamp tail
#pragma unroll
    for (int tau = 0; tau < 8; ++tau)
      pc[tau] = Pm[(size_t)t2 * G_DIM + tau * 16 + r];

    lstm_step(Bt, hrow, kb, lo, lane, pn, h, c);          // step t+1
  }

  hfin[m][lane] = h;
  __syncthreads();

  if (threadIdx.x == 0) {
    const float* hc = &hfin[0][0];      // concat(main,f1,f2,f3) row-major
    float a0 = bt[0];
    for (int k = 0; k < 4 * H_DIM; ++k) a0 += Wt[k] * hc[k];
    out[0] = a0;
    float a1 = bf1[0];
    for (int k = 0; k < H_DIM; ++k) a1 += Wf1[k] * hfin[1][k];
    out[1] = a1;
    float a2 = bf2[0];
    for (int k = 0; k < H_DIM; ++k) a2 += Wf2[k] * hfin[2][k];
    out[2] = a2;
    float a3 = bf2[0];                  // reference bug preserved: f3 uses Wf2/bf2
    for (int k = 0; k < H_DIM; ++k) a3 += Wf2[k] * hfin[3][k];
    out[3] = a3;
  }
}

extern "C" void kernel_launch(void* const* d_in, const int* in_sizes, int n_in,
                              void* d_out, int out_size, void* d_ws,
                              size_t ws_size, hipStream_t stream) {
  (void)in_sizes; (void)n_in; (void)out_size; (void)ws_size;
  const float* x    = (const float*)d_in[0];
  const float* W_ih = (const float*)d_in[1];
  const float* W_hh = (const float*)d_in[2];
  const float* b_ih = (const float*)d_in[3];
  const float* b_hh = (const float*)d_in[4];
  const float* Wt   = (const float*)d_in[5];
  const float* bt   = (const float*)d_in[6];
  const float* Wf1  = (const float*)d_in[7];
  const float* bf1  = (const float*)d_in[8];
  const float* Wf2  = (const float*)d_in[9];
  const float* bf2  = (const float*)d_in[10];

  float* P = (float*)d_ws;  // M*T*4H floats = 32 MiB scratch (fits in 192MB L2)

  // 4096 waves = 4 LSTMs x 1024 sixteen-step blocks
  lstm_xgate_precompute<<<1024, 128, 0, stream>>>(x, W_ih, b_ih, b_hh, P);
  // single workgroup, 4 waves, T sequential steps
  lstm_recurrence<<<1, 128, 0, stream>>>(P, W_hh, Wt, bt, Wf1, bf1, Wf2, bf2,
                                         (float*)d_out);
}
